// PointTransformerSequence_35347580846888
// MI455X (gfx1250) — compile-verified, matrix operand order checked by reading, not run on hardware
//
#include <hip/hip_runtime.h>

#define NPTS   50000
#define CCH    64
#define GRP    4
#define KNNS   16
#define DEPTHD 2
#define NTILES (NPTS / 16)          // 3125 16-row WMMA tiles
#define NSROWS (NPTS * KNNS)        // 800000

typedef _Float16 v16h __attribute__((ext_vector_type(16)));
typedef float    v8f  __attribute__((ext_vector_type(8)));

#define WAIT_LDS() asm volatile("s_wait_dscnt 0" ::: "memory")

// K-index held by (lane, element e) of a 16-bit WMMA A/B fragment (ISA 7.12.2)
__device__ __forceinline__ int wmma_kmap(int lane, int e) {
  int b = (e < 8) ? e : (8 + e);            // lanes 0-15: K = 0..7, 16..23
  return (lane < 16) ? b : (b + 8);         // lanes 16-31: K = 8..15, 24..31
}
__device__ __forceinline__ float relu_f(float x) { return x > 0.f ? x : 0.f; }

// ---------------------------------------------------------------------------
// Generic N x 64 @ 64 x 64 GEMM:  out = [relu(bn(in))] @ W + bias, + stats
// One wave = one 16-row tile, 4 col-tiles x 2 K-steps of v_wmma_f32_16x16x32_f16
// ---------------------------------------------------------------------------
__global__ __launch_bounds__(256) void gemm64_kernel(
    const float* __restrict__ in, const float* __restrict__ W,
    const float* __restrict__ bias,
    const float* __restrict__ bnMean, const float* __restrict__ bnRstd,
    const float* __restrict__ bnG, const float* __restrict__ bnB,
    float* __restrict__ out, float* __restrict__ stats)
{
  __shared__ _Float16 sW[64 * 64];
  __shared__ _Float16 sT[8][16 * 64];
  __shared__ float sStat[128];
  const int tid = threadIdx.x, lane = tid & 31, wid = tid >> 5;

  for (int i = tid; i < 4096; i += 256) sW[i] = (_Float16)W[i];
  if (tid < 128) sStat[tid] = 0.f;
  __syncthreads();

  const int tile = blockIdx.x * 8 + wid;
  if (tile < NTILES) {
    const int row0 = tile * 16;
    // stage 16x64 input tile as f16, fusing input BN + ReLU
    for (int v = lane; v < 256; v += 32) {
      const int r = v >> 4, c4 = (v & 15) * 4;
      const float4 x = *(const float4*)&in[(size_t)(row0 + r) * 64 + c4];
      float xv[4] = {x.x, x.y, x.z, x.w};
#pragma unroll
      for (int j = 0; j < 4; ++j) {
        float t = xv[j];
        if (bnMean) {
          const int ch = c4 + j;
          t = (t - bnMean[ch]) * bnRstd[ch] * bnG[ch] + bnB[ch];
          t = relu_f(t);
        }
        sT[wid][r * 64 + c4 + j] = (_Float16)t;
      }
    }
    WAIT_LDS();

    const int m = lane & 15;
    v16h a0, a1;
#pragma unroll
    for (int e = 0; e < 16; ++e) {
      const int k = wmma_kmap(lane, e);
      a0[e] = sT[wid][m * 64 + k];
      a1[e] = sT[wid][m * 64 + k + 32];
    }
#pragma unroll
    for (int nt = 0; nt < 4; ++nt) {
      const int n = nt * 16 + m;
      v16h b0, b1;
#pragma unroll
      for (int e = 0; e < 16; ++e) {
        const int k = wmma_kmap(lane, e);
        b0[e] = sW[k * 64 + n];
        b1[e] = sW[(k + 32) * 64 + n];
      }
      v8f c = {};
      c = __builtin_amdgcn_wmma_f32_16x16x32_f16(false, a0, false, b0, (short)0, c, false, false);
      c = __builtin_amdgcn_wmma_f32_16x16x32_f16(false, a1, false, b1, (short)0, c, false, false);
      const float bs = bias ? bias[n] : 0.f;
      float s1 = 0.f, s2 = 0.f;
#pragma unroll
      for (int i = 0; i < 8; ++i) {
        const float v = c[i] + bs;
        out[(size_t)(row0 + i + 8 * (lane >> 4)) * 64 + n] = v;
        s1 += v; s2 += v * v;
      }
      if (stats) {
        s1 += __shfl_xor(s1, 16, 32);
        s2 += __shfl_xor(s2, 16, 32);
        if (lane < 16) { atomicAdd(&sStat[n], s1); atomicAdd(&sStat[64 + n], s2); }
      }
    }
  }
  __syncthreads();
  if (stats && tid < 128) atomicAdd(&stats[tid], sStat[tid]);
}

// ---------------------------------------------------------------------------
// BN stat finalizer: acc = [sum(nch), sumsq(nch)] -> out = [mean(nch), rstd(nch)]
// ---------------------------------------------------------------------------
__global__ void finalize_kernel(const float* __restrict__ acc, float count,
                                float* __restrict__ outMR, int nch) {
  const int t = threadIdx.x;
  if (t < nch) {
    const float m = acc[t] / count;
    const float var = acc[nch + t] / count - m * m;
    outMR[t] = m;
    outMR[nch + t] = rsqrtf(var + 1e-5f);
  }
}

__global__ void zero_kernel(float* __restrict__ p, int n) {
  const int i = blockIdx.x * 256 + threadIdx.x;
  if (i < n) p[i] = 0.f;
}

// ---------------------------------------------------------------------------
// 9 second-order moments of pos=(coord[idx]-coord[n])*mask over all (n,s)
// -> exact BN stats of pos@pb1+b derived analytically (3x3 covariance trick)
// ---------------------------------------------------------------------------
__global__ __launch_bounds__(256) void posmom_kernel(const float* __restrict__ coord,
    const int* __restrict__ knn, float* __restrict__ mom) {
  __shared__ float sM[9];
  const int tid = threadIdx.x;
  if (tid < 9) sM[tid] = 0.f;
  __syncthreads();
  const int r = blockIdx.x * 256 + tid;          // grid sized exactly NSROWS/256
  const int n = r >> 4;
  const int kv = knn[r];
  const float mask = (kv >= 0) ? 1.f : 0.f;
  const int idx = (kv > 0) ? kv : 0;
  const float px = (coord[idx * 3 + 0] - coord[n * 3 + 0]) * mask;
  const float py = (coord[idx * 3 + 1] - coord[n * 3 + 1]) * mask;
  const float pz = (coord[idx * 3 + 2] - coord[n * 3 + 2]) * mask;
  float v[9] = {px, py, pz, px * px, py * py, pz * pz, px * py, px * pz, py * pz};
#pragma unroll
  for (int off = 16; off > 0; off >>= 1)
#pragma unroll
    for (int j = 0; j < 9; ++j) v[j] += __shfl_xor(v[j], off, 32);
  if ((tid & 31) == 0)
    for (int j = 0; j < 9; ++j) atomicAdd(&sM[j], v[j]);
  __syncthreads();
  if (tid < 9) atomicAdd(&mom[tid], sM[tid]);
}

__global__ void finpb_kernel(const float* __restrict__ mom,
    const float* __restrict__ pb1w, const float* __restrict__ pb1b,
    float* __restrict__ outMR) {
  const int ch = threadIdx.x;
  if (ch < 64) {
    const float inv = 1.f / (float)NSROWS;
    const float w0 = pb1w[ch], w1 = pb1w[64 + ch], w2 = pb1w[128 + ch], b = pb1b[ch];
    const float Epw = (mom[0] * w0 + mom[1] * w1 + mom[2] * w2) * inv;
    const float Epw2 = (mom[3] * w0 * w0 + mom[4] * w1 * w1 + mom[5] * w2 * w2 +
                        2.f * (mom[6] * w0 * w1 + mom[7] * w0 * w2 + mom[8] * w1 * w2)) * inv;
    const float m = Epw + b;
    const float var = (Epw2 + 2.f * b * Epw + b * b) - m * m;
    outMR[ch] = m;
    outMR[64 + ch] = rsqrtf(var + 1e-5f);
  }
}

// ---------------------------------------------------------------------------
// att1: one wave per point.  peb = relu(bn(pos@pb1+b)) @ pb2 + b2  (WMMA, S=16=M)
//       rel = bn_relu(k[idx])*mask - q + peb ;  w1 = rel @ we1 + b  -> W1 buffer
//       + global stats of w1 (G=4 channels)
// ---------------------------------------------------------------------------
__global__ __launch_bounds__(256) void att1_kernel(
    const float* __restrict__ coord, const int* __restrict__ knn,
    const float* __restrict__ yq, const float* __restrict__ yk,
    const float* __restrict__ qMR, const float* __restrict__ qG, const float* __restrict__ qBe,
    const float* __restrict__ kMR, const float* __restrict__ kG, const float* __restrict__ kBe,
    const float* __restrict__ pbMR, const float* __restrict__ pbG, const float* __restrict__ pbBe,
    const float* __restrict__ pb1w, const float* __restrict__ pb1b,
    const float* __restrict__ pb2w, const float* __restrict__ pb2b,
    const float* __restrict__ we1w, const float* __restrict__ we1b,
    float* __restrict__ w1out, float* __restrict__ statsWe)
{
  __shared__ _Float16 sPb2[4096];
  __shared__ float sPw[192], sScl[64], sSft[64];
  __shared__ float sKScl[64], sKSft[64], sQScl[64], sQSft[64];
  __shared__ float sWe1[256];
  __shared__ float sQ[8][64];
  __shared__ float sPosM[8][16][4];
  __shared__ int   sIdx[8][16];
  __shared__ float sRel[8][16 * 64];
  __shared__ float sStatW[8];
  const int tid = threadIdx.x, lane = tid & 31, wid = tid >> 5;

  for (int i = tid; i < 4096; i += 256) sPb2[i] = (_Float16)pb2w[i];
  for (int i = tid; i < 256; i += 256) sWe1[i] = we1w[i];
  if (tid < 192) sPw[tid] = pb1w[tid];
  if (tid < 64) {
    const float sc = pbMR[64 + tid] * pbG[tid];
    sScl[tid] = sc;
    sSft[tid] = (pb1b[tid] - pbMR[tid]) * sc + pbBe[tid];
    const float kc = kMR[64 + tid] * kG[tid];
    sKScl[tid] = kc; sKSft[tid] = kBe[tid] - kMR[tid] * kc;
    const float qc = qMR[64 + tid] * qG[tid];
    sQScl[tid] = qc; sQSft[tid] = qBe[tid] - qMR[tid] * qc;
  }
  if (tid < 8) sStatW[tid] = 0.f;
  __syncthreads();

  const int n = blockIdx.x * 8 + wid;          // grid is exactly NPTS/8
  if (lane < 16) {
    const int kv = knn[n * 16 + lane];
    const float mask = (kv >= 0) ? 1.f : 0.f;
    const int idx = (kv > 0) ? kv : 0;
    sIdx[wid][lane] = idx;
    const float cx = coord[n * 3], cy = coord[n * 3 + 1], cz = coord[n * 3 + 2];
    sPosM[wid][lane][0] = (coord[idx * 3 + 0] - cx) * mask;
    sPosM[wid][lane][1] = (coord[idx * 3 + 1] - cy) * mask;
    sPosM[wid][lane][2] = (coord[idx * 3 + 2] - cz) * mask;
    sPosM[wid][lane][3] = mask;
  }
  {
    const int c0 = 2 * lane;
    sQ[wid][c0]     = relu_f(yq[(size_t)n * 64 + c0]     * sQScl[c0]     + sQSft[c0]);
    sQ[wid][c0 + 1] = relu_f(yq[(size_t)n * 64 + c0 + 1] * sQScl[c0 + 1] + sQSft[c0 + 1]);
  }
  WAIT_LDS();

  // A fragments of relu(bn(pos@pb1+b)) built in-register (3-term dot per elem)
  const int m = lane & 15;
  const float px = sPosM[wid][m][0], py = sPosM[wid][m][1], pz = sPosM[wid][m][2];
  v16h a0, a1;
#pragma unroll
  for (int e = 0; e < 16; ++e) {
    const int k = wmma_kmap(lane, e);
    float t0 = px * sPw[k] + py * sPw[64 + k] + pz * sPw[128 + k];
    float t1 = px * sPw[k + 32] + py * sPw[96 + k] + pz * sPw[160 + k];
    a0[e] = (_Float16)relu_f(t0 * sScl[k] + sSft[k]);
    a1[e] = (_Float16)relu_f(t1 * sScl[k + 32] + sSft[k + 32]);
  }
#pragma unroll
  for (int nt = 0; nt < 4; ++nt) {
    const int nn = nt * 16 + m;
    v16h b0, b1;
#pragma unroll
    for (int e = 0; e < 16; ++e) {
      const int k = wmma_kmap(lane, e);
      b0[e] = sPb2[k * 64 + nn];
      b1[e] = sPb2[(k + 32) * 64 + nn];
    }
    v8f c = {};
    c = __builtin_amdgcn_wmma_f32_16x16x32_f16(false, a0, false, b0, (short)0, c, false, false);
    c = __builtin_amdgcn_wmma_f32_16x16x32_f16(false, a1, false, b1, (short)0, c, false, false);
    const float bb = pb2b[nn];
#pragma unroll
    for (int i = 0; i < 8; ++i)
      sRel[wid][(i + 8 * (lane >> 4)) * 64 + nn] = c[i] + bb;   // peb tile
  }
  WAIT_LDS();

  // rel = kg - q + peb   (lane owns channel pair 2*lane)
  const int c0 = 2 * lane, c1 = c0 + 1;
  const float qv0 = sQ[wid][c0], qv1 = sQ[wid][c1];
  for (int s = 0; s < 16; ++s) {
    const int idx = sIdx[wid][s];
    const float mask = sPosM[wid][s][3];
    if (s < 15) __builtin_prefetch(&yk[(size_t)sIdx[wid][s + 1] * 64 + c0], 0, 1);
    const float k0 = relu_f(yk[(size_t)idx * 64 + c0] * sKScl[c0] + sKSft[c0]) * mask;
    const float k1 = relu_f(yk[(size_t)idx * 64 + c1] * sKScl[c1] + sKSft[c1]) * mask;
    sRel[wid][s * 64 + c0] += k0 - qv0;
    sRel[wid][s * 64 + c1] += k1 - qv1;
  }
  WAIT_LDS();

  // w1 = rel @ we1 + b  : lane -> (s = lane&15, two g's)
  const int s = lane & 15;
  const int gb = (lane >> 4) * 2;
  float acc0 = we1b[gb], acc1 = we1b[gb + 1];
  for (int ch = 0; ch < 64; ++ch) {
    const float rv = sRel[wid][s * 64 + ch];
    acc0 += rv * sWe1[ch * 4 + gb];
    acc1 += rv * sWe1[ch * 4 + gb + 1];
  }
  w1out[((size_t)n * 16 + s) * 4 + gb]     = acc0;
  w1out[((size_t)n * 16 + s) * 4 + gb + 1] = acc1;
  atomicAdd(&sStatW[gb], acc0);         atomicAdd(&sStatW[4 + gb], acc0 * acc0);
  atomicAdd(&sStatW[gb + 1], acc1);     atomicAdd(&sStatW[4 + gb + 1], acc1 * acc1);
  __syncthreads();
  if (tid < 8) atomicAdd(&statsWe[tid], sStatW[tid]);
}

// ---------------------------------------------------------------------------
// att2: recompute peb (WMMA), w2 = relu(bn(w1)) @ we2 + b, softmax over S,
//       a = sum_s (v[idx]*mask + peb) * w   -> A buffer + n2 stats
// ---------------------------------------------------------------------------
__global__ __launch_bounds__(256) void att2_kernel(
    const float* __restrict__ coord, const int* __restrict__ knn,
    const float* __restrict__ vbuf, const float* __restrict__ w1buf,
    const float* __restrict__ pbMR, const float* __restrict__ pbG, const float* __restrict__ pbBe,
    const float* __restrict__ pb1w, const float* __restrict__ pb1b,
    const float* __restrict__ pb2w, const float* __restrict__ pb2b,
    const float* __restrict__ weMR, const float* __restrict__ weG, const float* __restrict__ weBe,
    const float* __restrict__ we2w, const float* __restrict__ we2b,
    float* __restrict__ aout, float* __restrict__ statsA)
{
  __shared__ _Float16 sPb2[4096];
  __shared__ float sPw[192], sScl[64], sSft[64];
  __shared__ float sWe2[16], sWScl[4], sWSft[4], sWe2b[4];
  __shared__ float sPosM[8][16][4];
  __shared__ int   sIdx[8][16];
  __shared__ float sVg[8][16 * 64];
  __shared__ float sWsm[8][16][4];
  __shared__ float sStatA[128];
  const int tid = threadIdx.x, lane = tid & 31, wid = tid >> 5;

  for (int i = tid; i < 4096; i += 256) sPb2[i] = (_Float16)pb2w[i];
  if (tid < 192) sPw[tid] = pb1w[tid];
  if (tid < 64) {
    const float sc = pbMR[64 + tid] * pbG[tid];
    sScl[tid] = sc;
    sSft[tid] = (pb1b[tid] - pbMR[tid]) * sc + pbBe[tid];
  }
  if (tid < 16) sWe2[tid] = we2w[tid];
  if (tid < 4) {
    const float sc = weMR[4 + tid] * weG[tid];
    sWScl[tid] = sc; sWSft[tid] = weBe[tid] - weMR[tid] * sc;
    sWe2b[tid] = we2b[tid];
  }
  if (tid < 128) sStatA[tid] = 0.f;
  __syncthreads();

  const int n = blockIdx.x * 8 + wid;
  if (lane < 16) {
    const int kv = knn[n * 16 + lane];
    const float mask = (kv >= 0) ? 1.f : 0.f;
    const int idx = (kv > 0) ? kv : 0;
    sIdx[wid][lane] = idx;
    const float cx = coord[n * 3], cy = coord[n * 3 + 1], cz = coord[n * 3 + 2];
    sPosM[wid][lane][0] = (coord[idx * 3 + 0] - cx) * mask;
    sPosM[wid][lane][1] = (coord[idx * 3 + 1] - cy) * mask;
    sPosM[wid][lane][2] = (coord[idx * 3 + 2] - cz) * mask;
    sPosM[wid][lane][3] = mask;
    // w2 = relu(bn(w1)) @ we2 + b
    const float4 w1v = *(const float4*)&w1buf[((size_t)n * 16 + lane) * 4];
    float hg[4] = {relu_f(w1v.x * sWScl[0] + sWSft[0]), relu_f(w1v.y * sWScl[1] + sWSft[1]),
                   relu_f(w1v.z * sWScl[2] + sWSft[2]), relu_f(w1v.w * sWScl[3] + sWSft[3])};
#pragma unroll
    for (int gp = 0; gp < 4; ++gp)
      sWsm[wid][lane][gp] = sWe2b[gp] + hg[0] * sWe2[gp] + hg[1] * sWe2[4 + gp] +
                            hg[2] * sWe2[8 + gp] + hg[3] * sWe2[12 + gp];
  }
  WAIT_LDS();
  if (lane < 4) {                       // softmax over S per group (masked after)
    float mx = -1e30f;
    for (int s = 0; s < 16; ++s) mx = fmaxf(mx, sWsm[wid][s][lane]);
    float den = 0.f;
    for (int s = 0; s < 16; ++s) { const float e = __expf(sWsm[wid][s][lane] - mx); sWsm[wid][s][lane] = e; den += e; }
    const float rd = 1.f / den;
    for (int s = 0; s < 16; ++s) sWsm[wid][s][lane] *= rd * sPosM[wid][s][3];
  }
  WAIT_LDS();

  // peb via WMMA into sVg
  const int m = lane & 15;
  const float px = sPosM[wid][m][0], py = sPosM[wid][m][1], pz = sPosM[wid][m][2];
  v16h a0, a1;
#pragma unroll
  for (int e = 0; e < 16; ++e) {
    const int k = wmma_kmap(lane, e);
    float t0 = px * sPw[k] + py * sPw[64 + k] + pz * sPw[128 + k];
    float t1 = px * sPw[k + 32] + py * sPw[96 + k] + pz * sPw[160 + k];
    a0[e] = (_Float16)relu_f(t0 * sScl[k] + sSft[k]);
    a1[e] = (_Float16)relu_f(t1 * sScl[k + 32] + sSft[k + 32]);
  }
#pragma unroll
  for (int nt = 0; nt < 4; ++nt) {
    const int nn = nt * 16 + m;
    v16h b0, b1;
#pragma unroll
    for (int e = 0; e < 16; ++e) {
      const int k = wmma_kmap(lane, e);
      b0[e] = sPb2[k * 64 + nn];
      b1[e] = sPb2[(k + 32) * 64 + nn];
    }
    v8f c = {};
    c = __builtin_amdgcn_wmma_f32_16x16x32_f16(false, a0, false, b0, (short)0, c, false, false);
    c = __builtin_amdgcn_wmma_f32_16x16x32_f16(false, a1, false, b1, (short)0, c, false, false);
    const float bb = pb2b[nn];
#pragma unroll
    for (int i = 0; i < 8; ++i)
      sVg[wid][(i + 8 * (lane >> 4)) * 64 + nn] = c[i] + bb;
  }
  WAIT_LDS();

  // aggregate: lane owns channels 2*lane, 2*lane+1 (same group)
  const int c0 = 2 * lane, c1 = c0 + 1;
  const int g = c0 >> 4;
  float acc0 = 0.f, acc1 = 0.f;
  for (int s = 0; s < 16; ++s) {
    const int idx = sIdx[wid][s];
    const float mask = sPosM[wid][s][3];
    if (s < 15) __builtin_prefetch(&vbuf[(size_t)sIdx[wid][s + 1] * 64 + c0], 0, 1);
    const float wv = sWsm[wid][s][g];
    acc0 += (vbuf[(size_t)idx * 64 + c0] * mask + sVg[wid][s * 64 + c0]) * wv;
    acc1 += (vbuf[(size_t)idx * 64 + c1] * mask + sVg[wid][s * 64 + c1]) * wv;
  }
  aout[(size_t)n * 64 + c0] = acc0;
  aout[(size_t)n * 64 + c1] = acc1;
  atomicAdd(&sStatA[c0], acc0);      atomicAdd(&sStatA[64 + c0], acc0 * acc0);
  atomicAdd(&sStatA[c1], acc1);      atomicAdd(&sStatA[64 + c1], acc1 * acc1);
  __syncthreads();
  if (tid < 128) atomicAdd(&statsA[tid], sStatA[tid]);
}

// x_next = relu(x + bn3(y3))
__global__ void resid_kernel(const float* __restrict__ xin, const float* __restrict__ y3,
    const float* __restrict__ mr, const float* __restrict__ g, const float* __restrict__ b,
    float* __restrict__ xout, int total) {
  const int i = blockIdx.x * 256 + threadIdx.x;
  if (i < total) {
    const int ch = i & 63;
    const float h = (y3[i] - mr[ch]) * mr[64 + ch] * g[ch] + b[ch];
    xout[i] = relu_f(xin[i] + h);
  }
}

// ---------------------------------------------------------------------------
extern "C" void kernel_launch(void* const* d_in, const int* in_sizes, int n_in,
                              void* d_out, int out_size, void* d_ws, size_t ws_size,
                              hipStream_t stream) {
  const float* feat  = (const float*)d_in[0];
  const float* coord = (const float*)d_in[1];
  const float* fc1_w = (const float*)d_in[2];
  const float* q_w = (const float*)d_in[3];  const float* q_b = (const float*)d_in[4];
  const float* q_g = (const float*)d_in[5];  const float* q_be = (const float*)d_in[6];
  const float* k_w = (const float*)d_in[7];  const float* k_b = (const float*)d_in[8];
  const float* k_g = (const float*)d_in[9];  const float* k_be = (const float*)d_in[10];
  const float* v_w = (const float*)d_in[11]; const float* v_b = (const float*)d_in[12];
  const float* pb1_w = (const float*)d_in[13]; const float* pb1_b = (const float*)d_in[14];
  const float* pb_g = (const float*)d_in[15];  const float* pb_be = (const float*)d_in[16];
  const float* pb2_w = (const float*)d_in[17]; const float* pb2_b = (const float*)d_in[18];
  const float* we1_w = (const float*)d_in[19]; const float* we1_b = (const float*)d_in[20];
  const float* we_g = (const float*)d_in[21];  const float* we_be = (const float*)d_in[22];
  const float* we2_w = (const float*)d_in[23]; const float* we2_b = (const float*)d_in[24];
  const float* fc3_w = (const float*)d_in[25];
  const float* n1_g = (const float*)d_in[26]; const float* n1_b = (const float*)d_in[27];
  const float* n2_g = (const float*)d_in[28]; const float* n2_b = (const float*)d_in[29];
  const float* n3_g = (const float*)d_in[30]; const float* n3_b = (const float*)d_in[31];
  const int* knn = (const int*)d_in[32];

  float* ws = (float*)d_ws;
  float* SB = ws;                                 // stats block (4096 floats)
  const size_t NC = (size_t)NPTS * 64;
  float* X0 = ws + 4096;
  float* Y1 = X0 + NC;  float* YQ = Y1 + NC;  float* YK = YQ + NC;
  float* Vb = YK + NC;  float* Ab = Vb + NC;  float* Y3 = Ab + NC;
  float* W1 = Y3 + NC;                            // N*S*G == NC floats

  // pos moments once (depth-invariant)
  zero_kernel<<<1, 256, 0, stream>>>(SB + 2048, 16);
  posmom_kernel<<<NSROWS / 256, 256, 0, stream>>>(coord, knn, SB + 2048);

  const float* Xin = feat;
  for (int d = 0; d < DEPTHD; ++d) {
    const int oCC = d * 4096, oC = d * 64, oPB = d * 192, oWE1 = d * 256, oG = d * 4, oW2 = d * 16;
    float* Xout = (d == DEPTHD - 1) ? (float*)d_out : X0;

    zero_kernel<<<4, 256, 0, stream>>>(SB, 1024);
    // fc1 (stats n1)
    gemm64_kernel<<<391, 256, 0, stream>>>(Xin, fc1_w + oCC, nullptr,
        nullptr, nullptr, nullptr, nullptr, Y1, SB + 0);
    finalize_kernel<<<1, 64, 0, stream>>>(SB + 0, (float)NPTS, SB + 1024, 64);
    // q / k / v from h = relu(bn1(y1))
    gemm64_kernel<<<391, 256, 0, stream>>>(Y1, q_w + oCC, q_b + oC,
        SB + 1024, SB + 1088, n1_g + oC, n1_b + oC, YQ, SB + 128);
    gemm64_kernel<<<391, 256, 0, stream>>>(Y1, k_w + oCC, k_b + oC,
        SB + 1024, SB + 1088, n1_g + oC, n1_b + oC, YK, SB + 256);
    gemm64_kernel<<<391, 256, 0, stream>>>(Y1, v_w + oCC, v_b + oC,
        SB + 1024, SB + 1088, n1_g + oC, n1_b + oC, Vb, nullptr);
    finalize_kernel<<<1, 64, 0, stream>>>(SB + 128, (float)NPTS, SB + 1152, 64);
    finalize_kernel<<<1, 64, 0, stream>>>(SB + 256, (float)NPTS, SB + 1280, 64);
    finpb_kernel<<<1, 64, 0, stream>>>(SB + 2048, pb1_w + oPB, pb1_b + oC, SB + 1408);
    // attention pass 1: w1 pre-activations + we stats
    att1_kernel<<<NPTS / 8, 256, 0, stream>>>(coord, knn, YQ, YK,
        SB + 1152, q_g + oC, q_be + oC,
        SB + 1280, k_g + oC, k_be + oC,
        SB + 1408, pb_g + oC, pb_be + oC,
        pb1_w + oPB, pb1_b + oC, pb2_w + oCC, pb2_b + oC,
        we1_w + oWE1, we1_b + oG, W1, SB + 512);
    finalize_kernel<<<1, 64, 0, stream>>>(SB + 512, (float)NSROWS, SB + 1536, 4);
    // attention pass 2: softmax + aggregation (stats n2)
    att2_kernel<<<NPTS / 8, 256, 0, stream>>>(coord, knn, Vb, W1,
        SB + 1408, pb_g + oC, pb_be + oC,
        pb1_w + oPB, pb1_b + oC, pb2_w + oCC, pb2_b + oC,
        SB + 1536, we_g + oG, we_be + oG, we2_w + oW2, we2_b + oG,
        Ab, SB + 640);
    finalize_kernel<<<1, 64, 0, stream>>>(SB + 640, (float)NPTS, SB + 1664, 64);
    // fc3 from relu(bn2(a)) (stats n3)
    gemm64_kernel<<<391, 256, 0, stream>>>(Ab, fc3_w + oCC, nullptr,
        SB + 1664, SB + 1728, n2_g + oC, n2_b + oC, Y3, SB + 768);
    finalize_kernel<<<1, 64, 0, stream>>>(SB + 768, (float)NPTS, SB + 1792, 64);
    // residual + relu
    resid_kernel<<<(int)((NC + 255) / 256), 256, 0, stream>>>(Xin, Y3,
        SB + 1792, n3_g + oC, n3_b + oC, Xout, (int)NC);
    Xin = Xout;
  }
  (void)in_sizes; (void)n_in; (void)out_size; (void)ws_size;
}
